// MiniMaxM2SparseMoeBlock_78752520339601
// MI455X (gfx1250) — matrix operand order repaired and checked by reading, not compiled, not run on hardware
//
#include <hip/hip_runtime.h>

// ---------------------------------------------------------------------------
// MiniMax-M2 sparse MoE block for MI455X (gfx1250), wave32 + WMMA bf16.
// T=4096, H=1536, I=768, E=64, K=8, C = 2*T*K/E = 1024.
// Weights ride the A operand (K-contiguous fragment), token tiles ride the
// B operand via a transposed LDS tile read with strided u16 loads.
// ---------------------------------------------------------------------------

#define MOE_T 4096
#define MOE_H 1536
#define MOE_I 768
#define MOE_E 64
#define MOE_K 8
#define MOE_C 1024

typedef __attribute__((ext_vector_type(16))) __bf16 v16bf;
typedef __attribute__((ext_vector_type(2)))  __bf16 v2bf;
typedef __attribute__((ext_vector_type(8)))  float  v8f;

// round-to-nearest-even fp32 -> bf16 (bits in [31:16])
__device__ __forceinline__ unsigned int rne_bf(unsigned int u) {
    return u + 0x7FFFu + ((u >> 16) & 1u);
}
__device__ __forceinline__ unsigned short f2bf(float x) {
    return (unsigned short)(rne_bf(__float_as_uint(x)) >> 16);
}

// pack two fp32 -> packed bf16x2 dword
#if __has_builtin(__builtin_amdgcn_cvt_pk_bf16_f32)
__device__ __forceinline__ unsigned int pack2(float x, float y) {
    auto r = __builtin_amdgcn_cvt_pk_bf16_f32(x, y);
    union { decltype(r) v; unsigned int u[(sizeof(r) + 3) / 4]; } cv;
    cv.v = r;
    return cv.u[0];
}
#else
__device__ __forceinline__ unsigned int pack2(float x, float y) {
    unsigned int ux = rne_bf(__float_as_uint(x));
    unsigned int uy = rne_bf(__float_as_uint(y));
    // result = [uy[31:16], ux[31:16]] : one v_perm_b32
    return __builtin_amdgcn_perm(uy, ux, 0x07060302u);
}
#endif

__device__ __forceinline__ float bf2f(unsigned short b) {
    return __uint_as_float(((unsigned int)b) << 16);
}

// ---------------------------------------------------------------------------
// 1) Router: logits = hs @ gate_w^T (fp32), sigmoid, top-8 (lowest-index tie
//    break to match jax.lax.top_k), weights renormalized from raw scores.
// ---------------------------------------------------------------------------
__global__ __launch_bounds__(64) void router_kernel(
    const float* __restrict__ hs,       // [T,H]
    const float* __restrict__ gate_w,   // [E,H]
    const float* __restrict__ bias,     // [E]
    int*   __restrict__ topk_idx,       // [T,K]
    float* __restrict__ topk_w)         // [T,K]
{
    const int t = blockIdx.x;
    const int e = threadIdx.x;

    const float4* x = (const float4*)(hs + (size_t)t * MOE_H);
    const float4* g = (const float4*)(gate_w + (size_t)e * MOE_H);
    float acc = 0.f;
    for (int q = 0; q < MOE_H / 4; ++q) {
        float4 a = x[q], b = g[q];
        acc += a.x * b.x + a.y * b.y + a.z * b.z + a.w * b.w;
    }
    float score = 1.0f / (1.0f + __expf(-acc));

    __shared__ float s_sel[MOE_E];
    __shared__ float s_score[MOE_E];
    s_sel[e]   = score + bias[e];
    s_score[e] = score;
    __syncthreads();

    if (e == 0) {
        unsigned long long used = 0ull;
        int   idxs[MOE_K];
        float wv[MOE_K];
        float wsum = 0.f;
        for (int k = 0; k < MOE_K; ++k) {
            int best = 0; float bv = -1e30f;
            for (int j = 0; j < MOE_E; ++j) {
                if ((used >> j) & 1ull) continue;
                float v = s_sel[j];
                if (v > bv) { bv = v; best = j; }
            }
            used |= (1ull << best);
            idxs[k] = best;
            wv[k]   = s_score[best];
            wsum   += wv[k];
        }
        float inv = 1.0f / wsum;
        for (int k = 0; k < MOE_K; ++k) {
            topk_idx[t * MOE_K + k] = idxs[k];
            topk_w[t * MOE_K + k]   = wv[k] * inv;
        }
    }
}

// ---------------------------------------------------------------------------
// 2) Dispatch: per-expert slot assignment.
// ---------------------------------------------------------------------------
__global__ void dispatch_kernel(
    const int* __restrict__ topk_idx,
    int* __restrict__ counts,
    int* __restrict__ row_map,
    int* __restrict__ dest)
{
    int i = blockIdx.x * blockDim.x + threadIdx.x;
    if (i >= MOE_T * MOE_K) return;
    int e = topk_idx[i];
    int pos = atomicAdd(&counts[e], 1);
    if (pos < MOE_C) {
        row_map[e * MOE_C + pos] = i;
        dest[i] = e * MOE_C + pos;
    } else {
        dest[i] = -1;
    }
}

// ---------------------------------------------------------------------------
// WMMA fragment helpers (documented gfx1250 wave32 VGPR layouts).
// A (16x32 bf16): lanes 0-15 -> M=lane, K in {0..7}U{16..23};
//                 lanes 16-31 -> M=lane-16, K in {8..15}U{24..31}.
// B (32x16 bf16): N=lane%16, K split by lane half exactly as for A.
// C (16x16 f32):  lane 0-15 -> N=lane, M=r; lane 16-31 -> N=lane-16, M=8+r.
// ---------------------------------------------------------------------------
#define LDSW 36    // weight tile K-stride (bf16 elems): 32 + 4 pad
#define LDXT 136   // transposed token tile N-stride: 128 + 8 pad (16B aligned)

// A fragment from a [rows][K] tile (K contiguous): 2x ds_load_2addr_b64.
__device__ __forceinline__ v16bf load_a_frag(const unsigned short* lW, int mBase, int lane) {
    v16bf a;
    const int m = lane & 15;
    const int half8 = (lane >> 4) << 3;   // 0 or 8
    const unsigned short* arow = lW + (mBase + m) * LDSW;
#pragma unroll
    for (int p = 0; p < 8; ++p) {
        int kk = (p < 4) ? (2 * p + half8) : (16 + 2 * (p - 4) + half8);
        unsigned int d = *(const unsigned int*)(arow + kk);
        v2bf pr = __builtin_bit_cast(v2bf, d);
        a[2 * p]     = pr[0];
        a[2 * p + 1] = pr[1];
    }
    return a;
}

// B fragment from the transposed token tile Xt[k][c]: per lane N=lane%16,
// elements walk K in table order (same as A): strided u16 LDS loads.
__device__ __forceinline__ v16bf load_bt_frag(const unsigned short* lXt, int nBase, int lane) {
    v16bf b;
    const int n = nBase + (lane & 15);
    const int half8 = (lane >> 4) << 3;
#pragma unroll
    for (int s = 0; s < 16; ++s) {
        int kk = (s < 8) ? (half8 + s) : (16 + half8 + (s - 8));
        b[s] = __builtin_bit_cast(__bf16, lXt[kk * LDXT + n]);
    }
    return b;
}

// ---------------------------------------------------------------------------
// 3) GEMM1 (fused gate+up+SiLU): Hh[c,i] = silu(X@Wg^T) * (X@Wu^T), bf16 out.
//    Block tile: 64 weight-outputs (M) x 128 token slots (N), K-step 32.
//    8 waves = 2(M) x 4(N), each 32x32 via 2x2 WMMA frags per matrix.
// ---------------------------------------------------------------------------
__global__ __launch_bounds__(256) void gemm1_kernel(
    const float* __restrict__ hs,       // [T,H]
    const float* __restrict__ w_gate,   // [E,I,H]
    const float* __restrict__ w_up,     // [E,I,H]
    const int*   __restrict__ row_map,  // [E*C]
    const int*   __restrict__ counts,   // [E]
    unsigned short* __restrict__ Hh)    // [E*C, I] bf16
{
    __shared__ __align__(16) unsigned short lWg[64 * LDSW];
    __shared__ __align__(16) unsigned short lWu[64 * LDSW];
    __shared__ __align__(16) unsigned short lXt[32 * LDXT];

    const int e       = blockIdx.y >> 3;
    const int rowBase = (blockIdx.y & 7) * 128;   // token tile
    const int iBase   = blockIdx.x * 64;          // weight-output tile
    const int cnt     = counts[e];
    if (rowBase >= cnt) return;   // block-uniform: EXEC stays all-ones

    const int tid  = threadIdx.x;
    const int lane = tid & 31;
    const int wid  = tid >> 5;
    const int wm   = wid & 1;     // weight strip (2 x 32)
    const int wn   = wid >> 1;    // token strip  (4 x 32)

    // weight stager: row n = tid>>2 (0..63), seg = tid&3 -> K cols seg*8..+8
    const int wrow = tid >> 2, wseg = tid & 3;
    const float* gptr = w_gate + ((size_t)e * MOE_I + (iBase + wrow)) * MOE_H + wseg * 8;
    const float* uptr = w_up   + ((size_t)e * MOE_I + (iBase + wrow)) * MOE_H + wseg * 8;

    // token stager: pair cp = tid>>2 (0..63) -> rows 2cp,2cp+1; kseg = tid&3
    const int cp = tid >> 2, kseg = tid & 3;
    const float* aptr0 = nullptr;
    const float* aptr1 = nullptr;
    {
        int rm0 = row_map[e * MOE_C + rowBase + 2 * cp];
        int rm1 = row_map[e * MOE_C + rowBase + 2 * cp + 1];
        if (rm0 >= 0) aptr0 = hs + (size_t)(rm0 >> 3) * MOE_H;
        if (rm1 >= 0) aptr1 = hs + (size_t)(rm1 >> 3) * MOE_H;
    }

    const float4 f4z = {0.f, 0.f, 0.f, 0.f};
    float4 gR[2], uR[2], x0R[2], x1R[2];
    auto stage_load = [&](int k0) {
        const float4* pg = (const float4*)(gptr + k0);
        gR[0] = pg[0]; gR[1] = pg[1];
        const float4* pu = (const float4*)(uptr + k0);
        uR[0] = pu[0]; uR[1] = pu[1];
        if (aptr0) { const float4* p = (const float4*)(aptr0 + k0 + kseg * 8); x0R[0] = p[0]; x0R[1] = p[1]; }
        else       { x0R[0] = f4z; x0R[1] = f4z; }
        if (aptr1) { const float4* p = (const float4*)(aptr1 + k0 + kseg * 8); x1R[0] = p[0]; x1R[1] = p[1]; }
        else       { x1R[0] = f4z; x1R[1] = f4z; }
    };

    v8f cg[2][2], cu[2][2];
#pragma unroll
    for (int i = 0; i < 2; ++i)
#pragma unroll
        for (int j = 0; j < 2; ++j)
#pragma unroll
            for (int r = 0; r < 8; ++r) { cg[i][j][r] = 0.f; cu[i][j][r] = 0.f; }

    stage_load(0);
    for (int k0 = 0; k0 < MOE_H; k0 += 32) {
        __syncthreads();
        {   // weights -> LDS (fp32->bf16, natural [row][K] layout)
            unsigned int* dg = (unsigned int*)&lWg[wrow * LDSW + wseg * 8];
            dg[0] = pack2(gR[0].x, gR[0].y); dg[1] = pack2(gR[0].z, gR[0].w);
            dg[2] = pack2(gR[1].x, gR[1].y); dg[3] = pack2(gR[1].z, gR[1].w);
            unsigned int* du = (unsigned int*)&lWu[wrow * LDSW + wseg * 8];
            du[0] = pack2(uR[0].x, uR[0].y); du[1] = pack2(uR[0].z, uR[0].w);
            du[2] = pack2(uR[1].x, uR[1].y); du[3] = pack2(uR[1].z, uR[1].w);
            // tokens -> LDS transposed: Xt[k][c0..c1] as packed pairs
            float a0[8] = {x0R[0].x, x0R[0].y, x0R[0].z, x0R[0].w,
                           x0R[1].x, x0R[1].y, x0R[1].z, x0R[1].w};
            float a1[8] = {x1R[0].x, x1R[0].y, x1R[0].z, x1R[0].w,
                           x1R[1].x, x1R[1].y, x1R[1].z, x1R[1].w};
#pragma unroll
            for (int q = 0; q < 8; ++q)
                *(unsigned int*)&lXt[(kseg * 8 + q) * LDXT + 2 * cp] = pack2(a0[q], a1[q]);
        }
        __syncthreads();
        if (k0 + 32 < MOE_H) stage_load(k0 + 32);   // prefetch next tile

        v16bf bx[2];
#pragma unroll
        for (int ni = 0; ni < 2; ++ni)
            bx[ni] = load_bt_frag(lXt, wn * 32 + ni * 16, lane);
#pragma unroll
        for (int mi = 0; mi < 2; ++mi) {
            v16bf ag = load_a_frag(lWg, wm * 32 + mi * 16, lane);
            v16bf au = load_a_frag(lWu, wm * 32 + mi * 16, lane);
#pragma unroll
            for (int ni = 0; ni < 2; ++ni) {
                cg[mi][ni] = __builtin_amdgcn_wmma_f32_16x16x32_bf16(
                    false, ag, false, bx[ni], (short)0, cg[mi][ni], false, false);
                cu[mi][ni] = __builtin_amdgcn_wmma_f32_16x16x32_bf16(
                    false, au, false, bx[ni], (short)0, cu[mi][ni], false, false);
            }
        }
    }

    // epilogue: lane carries token column, regs carry weight rows
    const int nt    = lane & 15;
    const int rbase = (lane >> 4) << 3;
    const size_t slot0 = (size_t)e * MOE_C + rowBase;
#pragma unroll
    for (int mi = 0; mi < 2; ++mi)
#pragma unroll
        for (int ni = 0; ni < 2; ++ni)
#pragma unroll
            for (int r = 0; r < 8; ++r) {
                float gv = cg[mi][ni][r];
                float uv = cu[mi][ni][r];
                float hv = (gv / (1.0f + __expf(-gv))) * uv;
                int c_row = wn * 32 + ni * 16 + nt;
                int i_col = iBase + wm * 32 + mi * 16 + rbase + r;
                Hh[(slot0 + c_row) * MOE_I + i_col] = f2bf(hv);
            }
}

// ---------------------------------------------------------------------------
// 4) GEMM2 (down-proj): Y[c,h] = Hh[c,:] @ Wd[h,:], bf16 out.
// ---------------------------------------------------------------------------
__global__ __launch_bounds__(256) void gemm2_kernel(
    const unsigned short* __restrict__ Hh,   // [E*C, I] bf16
    const float* __restrict__ w_down,        // [E,H,I]
    const int*   __restrict__ counts,
    unsigned short* __restrict__ Y)          // [E*C, H] bf16
{
    __shared__ __align__(16) unsigned short lW[64 * LDSW];
    __shared__ __align__(16) unsigned short lXt[32 * LDXT];

    const int e       = blockIdx.y >> 3;
    const int rowBase = (blockIdx.y & 7) * 128;
    const int hBase   = blockIdx.x * 64;
    const int cnt     = counts[e];
    if (rowBase >= cnt) return;

    const int tid  = threadIdx.x;
    const int lane = tid & 31;
    const int wid  = tid >> 5;
    const int wm   = wid & 1;
    const int wn   = wid >> 1;

    const int wrow = tid >> 2, wseg = tid & 3;
    const float* dptr = w_down + ((size_t)e * MOE_H + (hBase + wrow)) * MOE_I + wseg * 8;

    const int cp = tid >> 2, kseg = tid & 3;
    const size_t slot0 = (size_t)e * MOE_C + rowBase;
    const unsigned short* b0ptr = Hh + (slot0 + 2 * cp) * MOE_I + kseg * 8;
    const unsigned short* b1ptr = b0ptr + MOE_I;

    float4 dR[2];
    uint4  h0R, h1R;
    auto stage_load = [&](int k0) {
        const float4* pd = (const float4*)(dptr + k0);
        dR[0] = pd[0]; dR[1] = pd[1];
        h0R = *(const uint4*)(b0ptr + k0);
        h1R = *(const uint4*)(b1ptr + k0);
    };

    v8f c[2][2];
#pragma unroll
    for (int i = 0; i < 2; ++i)
#pragma unroll
        for (int j = 0; j < 2; ++j)
#pragma unroll
            for (int r = 0; r < 8; ++r) c[i][j][r] = 0.f;

    stage_load(0);
    for (int k0 = 0; k0 < MOE_I; k0 += 32) {
        __syncthreads();
        {   // weights -> LDS (fp32->bf16)
            unsigned int* d = (unsigned int*)&lW[wrow * LDSW + wseg * 8];
            d[0] = pack2(dR[0].x, dR[0].y); d[1] = pack2(dR[0].z, dR[0].w);
            d[2] = pack2(dR[1].x, dR[1].y); d[3] = pack2(dR[1].z, dR[1].w);
            // token rows (already bf16) -> transposed pair-packed LDS
            unsigned int a4[4] = {h0R.x, h0R.y, h0R.z, h0R.w};
            unsigned int b4[4] = {h1R.x, h1R.y, h1R.z, h1R.w};
#pragma unroll
            for (int q = 0; q < 4; ++q) {
                // lo = [b.lo16, a.lo16], hi = [b.hi16, a.hi16] : v_perm_b32
                unsigned int lo = __builtin_amdgcn_perm(b4[q], a4[q], 0x05040100u);
                unsigned int hi = __builtin_amdgcn_perm(b4[q], a4[q], 0x07060302u);
                *(unsigned int*)&lXt[(kseg * 8 + 2 * q) * LDXT + 2 * cp]     = lo;
                *(unsigned int*)&lXt[(kseg * 8 + 2 * q + 1) * LDXT + 2 * cp] = hi;
            }
        }
        __syncthreads();
        if (k0 + 32 < MOE_I) stage_load(k0 + 32);

        v16bf bx[2];
#pragma unroll
        for (int ni = 0; ni < 2; ++ni)
            bx[ni] = load_bt_frag(lXt, wn * 32 + ni * 16, lane);
#pragma unroll
        for (int mi = 0; mi < 2; ++mi) {
            v16bf a = load_a_frag(lW, wm * 32 + mi * 16, lane);
#pragma unroll
            for (int ni = 0; ni < 2; ++ni)
                c[mi][ni] = __builtin_amdgcn_wmma_f32_16x16x32_bf16(
                    false, a, false, bx[ni], (short)0, c[mi][ni], false, false);
        }
    }

    const int nt    = lane & 15;
    const int rbase = (lane >> 4) << 3;
#pragma unroll
    for (int mi = 0; mi < 2; ++mi)
#pragma unroll
        for (int ni = 0; ni < 2; ++ni)
#pragma unroll
            for (int r = 0; r < 8; ++r) {
                int c_row = wn * 32 + ni * 16 + nt;
                int h_col = hBase + wm * 32 + mi * 16 + rbase + r;
                Y[(slot0 + c_row) * MOE_H + h_col] = f2bf(c[mi][ni][r]);
            }
}

// ---------------------------------------------------------------------------
// 5) Combine: out[t,:] = sum_k w[t,k] * Y[dest[t,k],:]  (fixed k order ->
//    bit-deterministic regardless of dispatch atomics).
// ---------------------------------------------------------------------------
__global__ __launch_bounds__(256) void combine_kernel(
    const unsigned short* __restrict__ Y,    // [E*C, H] bf16
    const float* __restrict__ topk_w,        // [T,K]
    const int*   __restrict__ dest,          // [T*K]
    float* __restrict__ out)                 // [T,H]
{
    const int t   = blockIdx.x;
    const int tid = threadIdx.x;

    __shared__ float sw[MOE_K];
    __shared__ int   sd[MOE_K];
    if (tid < MOE_K) {
        sw[tid] = topk_w[t * MOE_K + tid];
        sd[tid] = dest[t * MOE_K + tid];
    }
    __syncthreads();

    // each thread handles 2 consecutive columns x 3 strided groups
    float2 acc[3];
#pragma unroll
    for (int j = 0; j < 3; ++j) { acc[j].x = 0.f; acc[j].y = 0.f; }

    for (int k = 0; k < MOE_K; ++k) {
        int d = sd[k];
        if (d < 0) continue;
        float wk = sw[k];
        const unsigned short* yrow = Y + (size_t)d * MOE_H;
#pragma unroll
        for (int j = 0; j < 3; ++j) {
            unsigned int u = *(const unsigned int*)&yrow[tid * 2 + 512 * j];
            acc[j].x += wk * bf2f((unsigned short)(u & 0xFFFFu));
            acc[j].y += wk * bf2f((unsigned short)(u >> 16));
        }
    }

    float* orow = out + (size_t)t * MOE_H;
#pragma unroll
    for (int j = 0; j < 3; ++j)
        *(float2*)&orow[tid * 2 + 512 * j] = acc[j];
}

// ---------------------------------------------------------------------------
// Host launcher
// ---------------------------------------------------------------------------
extern "C" void kernel_launch(void* const* d_in, const int* in_sizes, int n_in,
                              void* d_out, int out_size, void* d_ws, size_t ws_size,
                              hipStream_t stream) {
    (void)in_sizes; (void)n_in; (void)out_size; (void)ws_size;

    const float* hs     = (const float*)d_in[0];  // [T,H]
    const float* gate_w = (const float*)d_in[1];  // [E,H]
    const float* bias   = (const float*)d_in[2];  // [E]
    const float* w_gate = (const float*)d_in[3];  // [E,I,H]
    const float* w_up   = (const float*)d_in[4];  // [E,I,H]
    const float* w_down = (const float*)d_in[5];  // [E,H,I]
    float* out = (float*)d_out;

    char* ws = (char*)d_ws;
    size_t off = 0;
    auto take = [&](size_t bytes) -> char* {
        char* p = ws + off;
        off = (off + bytes + 255) & ~(size_t)255;
        return p;
    };
    int*   counts   = (int*)take(MOE_E * sizeof(int));
    int*   row_map  = (int*)take((size_t)MOE_E * MOE_C * sizeof(int));
    int*   topk_idx = (int*)take((size_t)MOE_T * MOE_K * sizeof(int));
    float* topk_w   = (float*)take((size_t)MOE_T * MOE_K * sizeof(float));
    int*   dest     = (int*)take((size_t)MOE_T * MOE_K * sizeof(int));
    unsigned short* Hh = (unsigned short*)take((size_t)MOE_E * MOE_C * MOE_I * 2);
    unsigned short* Yb = (unsigned short*)take((size_t)MOE_E * MOE_C * MOE_H * 2);

    hipMemsetAsync(counts, 0, MOE_E * sizeof(int), stream);
    hipMemsetAsync(row_map, 0xFF, (size_t)MOE_E * MOE_C * sizeof(int), stream);

    router_kernel<<<MOE_T, MOE_E, 0, stream>>>(hs, gate_w, bias, topk_idx, topk_w);
    dispatch_kernel<<<(MOE_T * MOE_K + 255) / 256, 256, 0, stream>>>(
        topk_idx, counts, row_map, dest);
    gemm1_kernel<<<dim3(MOE_I / 64, MOE_E * (MOE_C / 128)), 256, 0, stream>>>(
        hs, w_gate, w_up, row_map, counts, Hh);
    gemm2_kernel<<<dim3(MOE_H / 64, MOE_E * (MOE_C / 128)), 256, 0, stream>>>(
        Hh, w_down, counts, Yb);
    combine_kernel<<<MOE_T, 256, 0, stream>>>(Yb, topk_w, dest, out);
}
